// ModelDisigner_8555574854414
// MI455X (gfx1250) — matrix operand order repaired and checked by reading, not compiled, not run on hardware
//
#include <hip/hip_runtime.h>
#include <math.h>

typedef __attribute__((ext_vector_type(16))) _Float16 v16h;
typedef __attribute__((ext_vector_type(8)))  _Float16 v8h;
typedef __attribute__((ext_vector_type(8)))  float    v8f;
typedef int v4i_g __attribute__((vector_size(16)));   // matches builtin param shape

// Optional CDNA5 async global->LDS staging (compile-safe feature probe)
#if defined(__has_builtin)
#  if __has_builtin(__builtin_amdgcn_global_load_async_to_lds_b128) && \
      __has_builtin(__builtin_amdgcn_s_wait_asynccnt)
#    define USE_ASYNC_LDS 1
#  endif
#endif

#define GLOBAL_AS __attribute__((address_space(1)))
#define LDS_AS    __attribute__((address_space(3)))

// ---------------------------------------------------------------------------
// Weight repack: fp32 [Cout][K] (OIHW flat) -> fp16 [Cout][Kpad], zero padded
// ---------------------------------------------------------------------------
__global__ __launch_bounds__(256)
void k_cvt_pad(const float* __restrict__ in, _Float16* __restrict__ out,
               int Cout, int K, int Kpad)
{
    const int total = Cout * Kpad;
    int i = blockIdx.x * 256 + threadIdx.x;
    if (i >= total) return;
    const int oc = i / Kpad, k = i % Kpad;
    out[i] = (k < K) ? (_Float16)in[(size_t)oc * K + k] : (_Float16)0.f;
}

// mask_w fp32 [256][8192] -> fp16 [8192][256] (transpose so FC uses fast path)
__global__ __launch_bounds__(256)
void k_cvt_maskw(const float* __restrict__ in, _Float16* __restrict__ out)
{
    const int total = 8192 * 256;
    int i = blockIdx.x * 256 + threadIdx.x;
    if (i >= total) return;
    const int oc = i / 256, k = i % 256;
    out[i] = (_Float16)in[(size_t)k * 8192 + oc];
}

// ---------------------------------------------------------------------------
// Conv (k=1 or k=3 pad=1) as LDS-staged implicit GEMM on WMMA f16->f32.
//   Block = 256 threads (8 waves), macro-tile BM=64 x BN=64, K-chunk 64.
//   Waves 2(M)x4(N); each wave owns 32x16 => 4 WMMA per stage (2 M x 2 Kstep).
//   LDS rows padded to 72 halves (144B): 16B aligned, 36-word bank stride.
//   B tile staged via GLOBAL_LOAD_ASYNC_TO_LDS_B128 (ASYNCcnt) when available.
// ---------------------------------------------------------------------------
#define BM 64
#define BN 64
#define KC 64
#define LDT 72

__global__ __launch_bounds__(256)
void k_conv_wmma(const float* __restrict__ in, const _Float16* __restrict__ wgt,
                 const float* __restrict__ bias, float* __restrict__ out,
                 int Cin, int Cout, int H, int W, int ksz, int Kpad, int relu)
{
    __shared__ _Float16 lA[BM * LDT];
    __shared__ _Float16 lB[BN * LDT];

    const int M = H * W;
    const int K = Cin * ksz * ksz;
    const int tid  = threadIdx.x;
    const int lane = tid & 31;
    const int wv   = tid >> 5;
    const int wn   = wv & 3;          // N sub-tile 0..3
    const int wm   = wv >> 2;         // M sub-pair 0..1
    const int lr   = lane & 15;
    const int hg   = lane >> 4;       // K-half group per ISA fragment layout

    const int mb  = blockIdx.x * BM;
    const int ocb = blockIdx.y * BN;
    const int n   = blockIdx.z;

    // staging coordinates: thread -> (row 0..63, 16-half segment 0..3)
    const int frow = tid >> 2;
    const int fseg = tid & 3;
    const int fm     = mb + frow;
    const int fvalid = (fm < M);
    const int foy = fvalid ? fm / W : 0;
    const int fox = fvalid ? fm % W : 0;
    const int fn  = ocb + frow;

    const float*    __restrict__ inN  = in  + (size_t)n * Cin * M;
    const _Float16* __restrict__ wrow = wgt + (size_t)fn * Kpad + fseg * 16;

    v8f acc0 = {0.f, 0.f, 0.f, 0.f, 0.f, 0.f, 0.f, 0.f};
    v8f acc1 = acc0;

    for (int kc = 0; kc < K; kc += KC) {
        __syncthreads();
        // ---- stage B tile (weights, padded f16: guard-free segments) ----
        {
            _Float16* dstB = &lB[frow * LDT + fseg * 16];
            if (fn < Cout) {
#ifdef USE_ASYNC_LDS
                __builtin_amdgcn_global_load_async_to_lds_b128(
                    (GLOBAL_AS v4i_g*)(wrow + kc),
                    (LDS_AS v4i_g*)dstB, 0, 0);
                __builtin_amdgcn_global_load_async_to_lds_b128(
                    (GLOBAL_AS v4i_g*)(wrow + kc + 8),
                    (LDS_AS v4i_g*)(dstB + 8), 0, 0);
#else
                *(uint4*)dstB       = *(const uint4*)(wrow + kc);
                *(uint4*)(dstB + 8) = *(const uint4*)(wrow + kc + 8);
#endif
            } else {
                uint4 z = {0u, 0u, 0u, 0u};
                *(uint4*)dstB       = z;
                *(uint4*)(dstB + 8) = z;
            }
        }
        if (kc + KC < K) __builtin_prefetch(wrow + kc + KC, 0, 3);
        // ---- stage A tile: im2col gather, f32 -> f16 ----
        {
            alignas(16) _Float16 tmp[16];
            if (ksz == 1) {
#pragma unroll
                for (int j = 0; j < 16; ++j) {
                    const int k = kc + fseg * 16 + j;
                    float v = (fvalid && k < Cin) ? inN[(size_t)k * M + fm] : 0.f;
                    tmp[j] = (_Float16)v;
                }
            } else {            // 3x3, pad 1: constant divisors -> mul/shift
#pragma unroll
                for (int j = 0; j < 16; ++j) {
                    const int k = kc + fseg * 16 + j;
                    float v = 0.f;
                    if (fvalid && k < K) {
                        const int ci = k / 9, rem = k - ci * 9;
                        const int ky = rem / 3, kx = rem - ky * 3;
                        const int iy = foy + ky - 1, ix = fox + kx - 1;
                        if ((unsigned)iy < (unsigned)H && (unsigned)ix < (unsigned)W)
                            v = inN[(size_t)ci * M + iy * W + ix];
                    }
                    tmp[j] = (_Float16)v;
                }
            }
            _Float16* dstA = &lA[frow * LDT + fseg * 16];
            *(uint4*)dstA       = *(const uint4*)tmp;
            *(uint4*)(dstA + 8) = *(const uint4*)(tmp + 8);
        }
#ifdef USE_ASYNC_LDS
        __builtin_amdgcn_s_wait_asynccnt(0);
#endif
        __syncthreads();
        // ---- fragments from LDS (ds_load_b128), 4 WMMA per stage ----
        const _Float16* pa0 = &lA[(wm * 32 + lr) * LDT];
        const _Float16* pa1 = pa0 + 16 * LDT;
        const _Float16* pb  = &lB[(wn * 16 + lr) * LDT];
#pragma unroll
        for (int q = 0; q < 2; ++q) {
            const int ko = q * 32;
            const v8h a0lo = *(const v8h*)(pa0 + ko + hg * 8);
            const v8h a0hi = *(const v8h*)(pa0 + ko + 16 + hg * 8);
            const v8h a1lo = *(const v8h*)(pa1 + ko + hg * 8);
            const v8h a1hi = *(const v8h*)(pa1 + ko + 16 + hg * 8);
            const v8h blo  = *(const v8h*)(pb + ko + hg * 8);
            const v8h bhi  = *(const v8h*)(pb + ko + 16 + hg * 8);
            const v16h a0 = __builtin_shufflevector(a0lo, a0hi, 0,1,2,3,4,5,6,7,8,9,10,11,12,13,14,15);
            const v16h a1 = __builtin_shufflevector(a1lo, a1hi, 0,1,2,3,4,5,6,7,8,9,10,11,12,13,14,15);
            const v16h bf = __builtin_shufflevector(blo,  bhi,  0,1,2,3,4,5,6,7,8,9,10,11,12,13,14,15);
            acc0 = __builtin_amdgcn_wmma_f32_16x16x32_f16(false, a0, false, bf, (short)0, acc0, false, false);
            acc1 = __builtin_amdgcn_wmma_f32_16x16x32_f16(false, a1, false, bf, (short)0, acc1, false, false);
        }
    }

    // ---- epilogue: bias + ReLU; vector stores when M is a full-tile multiple
    const int ncol = ocb + wn * 16 + lr;
    if (ncol < Cout) {
        const float bb = bias ? bias[ncol] : 0.f;
        float* __restrict__ orow = out + ((size_t)n * Cout + ncol) * M;
        const int mbase = mb + wm * 32 + hg * 8;
        if ((M & (BM - 1)) == 0) {   // all conv layers except 17x17 head & FC
            float r0[8], r1[8];
#pragma unroll
            for (int r = 0; r < 8; ++r) {
                float x0 = acc0[r] + bb, x1 = acc1[r] + bb;
                if (relu) { x0 = fmaxf(x0, 0.f); x1 = fmaxf(x1, 0.f); }
                r0[r] = x0; r1[r] = x1;
            }
            *(float4*)(orow + mbase)          = make_float4(r0[0], r0[1], r0[2], r0[3]);
            *(float4*)(orow + mbase + 4)      = make_float4(r0[4], r0[5], r0[6], r0[7]);
            *(float4*)(orow + mbase + 16)     = make_float4(r1[0], r1[1], r1[2], r1[3]);
            *(float4*)(orow + mbase + 16 + 4) = make_float4(r1[4], r1[5], r1[6], r1[7]);
        } else {
#pragma unroll
            for (int r = 0; r < 8; ++r) {
                const int m0 = mbase + r;
                if (m0 < M) {
                    float v = acc0[r] + bb;
                    if (relu) v = fmaxf(v, 0.f);
                    orow[m0] = v;
                }
                const int m1 = m0 + 16;
                if (m1 < M) {
                    float v = acc1[r] + bb;
                    if (relu) v = fmaxf(v, 0.f);
                    orow[m1] = v;
                }
            }
        }
    }
}

// ---------------------------------------------------------------------------
// 2x2 max pool, stride 2
// ---------------------------------------------------------------------------
__global__ __launch_bounds__(256)
void k_pool2(const float* __restrict__ in, float* __restrict__ out,
             int NC, int H, int W)
{
    const int Ho = H >> 1, Wo = W >> 1;
    const int total = NC * Ho * Wo;
    int i = blockIdx.x * 256 + threadIdx.x;
    if (i >= total) return;
    const int x = i % Wo; int t = i / Wo;
    const int y = t % Ho; const int c = t / Ho;
    const float* p = in + ((size_t)c * H + 2 * y) * W + 2 * x;
    out[i] = fmaxf(fmaxf(p[0], p[1]), fmaxf(p[W], p[W + 1]));
}

// nearest upsample x2 written into channel-slice [0,C) of a Cdst-channel dst
__global__ __launch_bounds__(256)
void k_up2_cat(const float* __restrict__ src, float* __restrict__ dst,
               int N, int C, int Hs, int Ws, int Cdst)
{
    const int Ho = Hs * 2, Wo = Ws * 2;
    const int total = N * C * Ho * Wo;
    int i = blockIdx.x * 256 + threadIdx.x;
    if (i >= total) return;
    const int x = i % Wo; int t = i / Wo;
    const int y = t % Ho; t /= Ho;
    const int c = t % C;  const int n = t / C;
    dst[(((size_t)n * Cdst + c) * Ho + y) * Wo + x] =
        src[(((size_t)n * C + c) * Hs + (y >> 1)) * Ws + (x >> 1)];
}

// crop 16*scale window at scale*pos, write into channel-slice [choff, choff+C)
__global__ __launch_bounds__(256)
void k_crop_cat(const float* __restrict__ src, float* __restrict__ dst,
                const int* __restrict__ pos,
                int N, int Csrc, int Hsrc, int scale, int Cdst, int choff)
{
    const int sz = 16 * scale;
    const int total = N * Csrc * sz * sz;
    int i = blockIdx.x * 256 + threadIdx.x;
    if (i >= total) return;
    const int x = i % sz; int t = i / sz;
    const int y = t % sz; t /= sz;
    const int c = t % Csrc; const int n = t / Csrc;
    const int py = pos[2 * n] * scale, px = pos[2 * n + 1] * scale;
    dst[(((size_t)n * Cdst + choff + c) * sz + y) * sz + x] =
        src[(((size_t)n * Csrc + c) * Hsrc + py + y) * Hsrc + px + x];
}

// mfeat [N,8192] -> channels 0..31 of cat4 [N,544,16,16]
__global__ __launch_bounds__(256)
void k_mfeat_cat(const float* __restrict__ mf, float* __restrict__ dst, int N)
{
    const int total = N * 32 * 256;
    int i = blockIdx.x * 256 + threadIdx.x;
    if (i >= total) return;
    const int n = i / 8192, r = i % 8192;
    dst[(size_t)n * 544 * 256 + r] = mf[(size_t)n * 8192 + r];
}

// depthwise cross-correlation: corr[n][c][oy][ox] =
//   sum_{ky,kx<16} sfeat[n][c][oy+ky][ox+kx] * tfeat[n%B][c][ky][kx]
__global__ __launch_bounds__(256)
void k_xcorr(const float* __restrict__ sf, const float* __restrict__ tf,
             float* __restrict__ corr, int N, int Bsz)
{
    const int total = N * 256 * 289;
    int i = blockIdx.x * 256 + threadIdx.x;
    if (i >= total) return;
    const int p = i % 289; int t = i / 289;
    const int c = t % 256; const int n = t / 256;
    const int oy = p / 17, ox = p % 17;
    const int b = n % Bsz;
    const float* __restrict__ S  = sf + (((size_t)n * 256 + c) << 10); // 32*32
    const float* __restrict__ Tm = tf + (((size_t)b * 256 + c) << 8);  // 16*16
    float s = 0.f;
#pragma unroll 4
    for (int ky = 0; ky < 16; ++ky)
        for (int kx = 0; kx < 16; ++kx)
            s += S[(oy + ky) * 32 + ox + kx] * Tm[ky * 16 + kx];
    corr[i] = s;
}

// per-image argmax over 17x17 score map (first max, matching jnp.argmax)
__global__ __launch_bounds__(32)
void k_argmax(const float* __restrict__ score, int* __restrict__ pos, int N)
{
    const int n = threadIdx.x;
    if (n >= N) return;
    const float* s = score + (size_t)n * 289;
    float best = s[0]; int bi = 0;
    for (int i = 1; i < 289; ++i) if (s[i] > best) { best = s[i]; bi = i; }
    pos[2 * n]     = bi / 17;
    pos[2 * n + 1] = bi % 17;
}

// row[n][c] = corr[n][c][py][px]
__global__ __launch_bounds__(256)
void k_gather_row(const float* __restrict__ corr, const int* __restrict__ pos,
                  float* __restrict__ row, int N)
{
    const int total = N * 256;
    int i = blockIdx.x * 256 + threadIdx.x;
    if (i >= total) return;
    const int n = i / 256;
    row[i] = corr[(size_t)i * 289 + pos[2 * n] * 17 + pos[2 * n + 1]];
}

// searchs [B,T,3,256,256] -> s [T*B,3,256,256]
__global__ __launch_bounds__(256)
void k_tr_searchs(const float* __restrict__ s, float* __restrict__ d,
                  int Bsz, int Tn)
{
    const int per = 3 * 256 * 256;
    const int total = Bsz * Tn * per;
    int i = blockIdx.x * 256 + threadIdx.x;
    if (i >= total) return;
    const int bt = i / per, r = i % per;
    const int b = bt / Tn, t = bt % Tn;
    d[((size_t)(t * Bsz + b)) * per + r] = s[i];
}

// score [N,289] -> out [B,T,1,17,17]
__global__ __launch_bounds__(256)
void k_out_score(const float* __restrict__ sc, float* __restrict__ out,
                 int Bsz, int Tn)
{
    const int total = Bsz * Tn * 289;
    int i = blockIdx.x * 256 + threadIdx.x;
    if (i >= total) return;
    const int bt = i / 289, r = i % 289;
    const int b = bt / Tn, t = bt % Tn;
    out[i] = sc[(size_t)(t * Bsz + b) * 289 + r];
}

// fin [N,1,128,128] -> sigmoid -> out [B,T,1,128,128]
__global__ __launch_bounds__(256)
void k_out_mask(const float* __restrict__ S, float* __restrict__ out,
                int Bsz, int Tn)
{
    const int total = Bsz * Tn * 16384;
    int i = blockIdx.x * 256 + threadIdx.x;
    if (i >= total) return;
    const int bt = i / 16384, r = i % 16384;
    const int b = bt / Tn, t = bt % Tn;
    const float v = S[(size_t)(t * Bsz + b) * 16384 + r];
    out[i] = 1.f / (1.f + expf(-v));
}

// ---------------------------------------------------------------------------
// Layer table: {Cout, Cin, k} in _CONV_SPECS order (NC = 1)
// ---------------------------------------------------------------------------
static const int SPEC[20][3] = {
    {64, 3, 3},   {64, 64, 3},  {128, 64, 3},  {128, 128, 3},
    {256, 128, 3},{256, 256, 3},{512, 256, 3}, {512, 512, 3},
    {256, 512, 1},{256, 256, 1},{1, 256, 1},
    {512, 544, 3},{512, 512, 3},{256, 768, 3}, {256, 256, 3},
    {128, 384, 3},{128, 128, 3},{64, 192, 3},  {64, 64, 3},  {1, 64, 1}
};

extern "C" void kernel_launch(void* const* d_in, const int* in_sizes, int n_in,
                              void* d_out, int out_size, void* d_ws, size_t ws_size,
                              hipStream_t stream)
{
    (void)in_sizes; (void)n_in; (void)out_size; (void)ws_size;
    const int Bsz = 2, Tn = 4, N = 8;

    const float* target  = (const float*)d_in[0];
    const float* searchs = (const float*)d_in[1];
    auto Wp = [&](int j) { return (const float*)d_in[2 + 2 * j]; };
    auto Bp = [&](int j) { return (const float*)d_in[3 + 2 * j]; };
    const float* mask_b = (const float*)d_in[43];

    // ---- bump allocator over workspace ----
    char* ws = (char*)d_ws;
    size_t cur = 0;
    auto ALLOC = [&](size_t bytes) -> void* {
        void* p = ws + cur;
        cur += (bytes + 255) & ~(size_t)255;
        return p;
    };
    auto AF = [&](size_t n) { return (float*)ALLOC(n * sizeof(float)); };
    auto AH = [&](size_t n) { return (_Float16*)ALLOC(n * sizeof(_Float16)); };

    // ---- repack all weights to padded fp16 arena (Kpad 64-aligned) ----
    _Float16* wgt[21];
    int kpad[21];
    for (int j = 0; j < 20; ++j) {
        const int K  = SPEC[j][1] * SPEC[j][2] * SPEC[j][2];
        const int Kp = (K + 63) & ~63;
        const size_t cnt = (size_t)SPEC[j][0] * Kp;
        kpad[j] = Kp;
        wgt[j]  = AH(cnt);
        k_cvt_pad<<<(int)((cnt + 255) / 256), 256, 0, stream>>>(Wp(j), wgt[j], SPEC[j][0], K, Kp);
    }
    {   // mask_w -> transposed [8192][256]
        const size_t cnt = (size_t)8192 * 256;
        kpad[20] = 256;
        wgt[20]  = AH(cnt);
        k_cvt_maskw<<<(int)((cnt + 255) / 256), 256, 0, stream>>>((const float*)d_in[42], wgt[20]);
    }

    // ---- activation buffers ----
    float* sx    = AF((size_t)N * 3 * 65536);
    float* keep1 = AF((size_t)N * 64 * 65536);   // d1 (search)
    float* keep2 = AF((size_t)N * 128 * 16384);  // d2
    float* keep3 = AF((size_t)N * 256 * 4096);   // d3
    float* keep4 = AF((size_t)N * 512 * 1024);   // d4
    float* sfeat = AF((size_t)N * 256 * 1024);
    float* tfeat = AF((size_t)Bsz * 256 * 256);
    float* corr  = AF((size_t)N * 256 * 289);
    float* hbuf  = AF((size_t)N * 256 * 289);
    float* score = AF((size_t)N * 289);
    int*   pos   = (int*)ALLOC((size_t)N * 2 * sizeof(int));
    float* rowv  = AF((size_t)N * 256);
    float* mfeat = AF((size_t)N * 8192);
    float* cat4  = AF((size_t)N * 544 * 256);
    float* A4    = AF((size_t)N * 512 * 256);
    float* B4    = AF((size_t)N * 512 * 256);
    float* cat3  = AF((size_t)N * 768 * 1024);
    float* A3    = AF((size_t)N * 256 * 1024);
    float* B3    = AF((size_t)N * 256 * 1024);
    float* cat2  = AF((size_t)N * 384 * 4096);
    float* A2    = AF((size_t)N * 128 * 4096);
    float* B2    = AF((size_t)N * 128 * 4096);
    float* cat1  = AF((size_t)N * 192 * 16384);
    float* A1    = AF((size_t)N * 64 * 16384);
    float* B1    = AF((size_t)N * 64 * 16384);
    float* fin   = AF((size_t)N * 16384);
    float* tmpA  = AF((size_t)N * 64 * 65536);   // 33.5M floats
    float* tmpB  = AF((size_t)N * 64 * 16384);   // 8.4M floats

    // ---- launch helpers ----
    auto conv = [&](const float* in, float* out, int j, const float* b,
                    int Nb, int Cin, int Cout, int H, int Wd, int k, int relu) {
        dim3 grid((unsigned)((H * Wd + BM - 1) / BM), (unsigned)((Cout + BN - 1) / BN), (unsigned)Nb);
        k_conv_wmma<<<grid, 256, 0, stream>>>(in, wgt[j], b, out, Cin, Cout, H, Wd, k, kpad[j], relu);
    };
    auto pool = [&](const float* in, float* out, int NC, int H, int Wd) {
        const int tot = NC * (H / 2) * (Wd / 2);
        k_pool2<<<(tot + 255) / 256, 256, 0, stream>>>(in, out, NC, H, Wd);
    };

    // ================= target backbone (B images, 128x128) =================
    conv(target, tmpA, 0, Bp(0), Bsz, 3, 64, 128, 128, 3, 1);
    conv(tmpA, tmpB, 1, Bp(1), Bsz, 64, 64, 128, 128, 3, 1);
    pool(tmpB, tmpA, Bsz * 64, 128, 128);
    conv(tmpA, tmpB, 2, Bp(2), Bsz, 64, 128, 64, 64, 3, 1);
    conv(tmpB, tmpA, 3, Bp(3), Bsz, 128, 128, 64, 64, 3, 1);
    pool(tmpA, tmpB, Bsz * 128, 64, 64);
    conv(tmpB, tmpA, 4, Bp(4), Bsz, 128, 256, 32, 32, 3, 1);
    conv(tmpA, tmpB, 5, Bp(5), Bsz, 256, 256, 32, 32, 3, 1);
    pool(tmpB, tmpA, Bsz * 256, 32, 32);
    conv(tmpA, tmpB, 6, Bp(6), Bsz, 256, 512, 16, 16, 3, 1);
    conv(tmpB, tmpA, 7, Bp(7), Bsz, 512, 512, 16, 16, 3, 1);
    conv(tmpA, tfeat, 8, Bp(8), Bsz, 512, 256, 16, 16, 1, 0);

    // ================= search backbone (N=8 images, 256x256) ================
    {
        const int tot = Bsz * Tn * 3 * 65536;
        k_tr_searchs<<<(tot + 255) / 256, 256, 0, stream>>>(searchs, sx, Bsz, Tn);
    }
    conv(sx, tmpA, 0, Bp(0), N, 3, 64, 256, 256, 3, 1);
    conv(tmpA, keep1, 1, Bp(1), N, 64, 64, 256, 256, 3, 1);
    pool(keep1, tmpB, N * 64, 256, 256);
    conv(tmpB, tmpA, 2, Bp(2), N, 64, 128, 128, 128, 3, 1);
    conv(tmpA, keep2, 3, Bp(3), N, 128, 128, 128, 128, 3, 1);
    pool(keep2, tmpB, N * 128, 128, 128);
    conv(tmpB, tmpA, 4, Bp(4), N, 128, 256, 64, 64, 3, 1);
    conv(tmpA, keep3, 5, Bp(5), N, 256, 256, 64, 64, 3, 1);
    pool(keep3, tmpB, N * 256, 64, 64);
    conv(tmpB, tmpA, 6, Bp(6), N, 256, 512, 32, 32, 3, 1);
    conv(tmpA, keep4, 7, Bp(7), N, 512, 512, 32, 32, 3, 1);
    conv(keep4, sfeat, 8, Bp(8), N, 512, 256, 32, 32, 1, 0);

    // ================= xcorr + score head + argmax ==========================
    {
        const int tot = N * 256 * 289;
        k_xcorr<<<(tot + 255) / 256, 256, 0, stream>>>(sfeat, tfeat, corr, N, Bsz);
    }
    conv(corr, hbuf, 9, Bp(9), N, 256, 256, 17, 17, 1, 1);
    conv(hbuf, score, 10, Bp(10), N, 256, 1, 17, 17, 1, 0);
    k_argmax<<<1, 32, 0, stream>>>(score, pos, N);
    k_gather_row<<<(N * 256 + 255) / 256, 256, 0, stream>>>(corr, pos, rowv, N);
    // mask FC: [N,256] @ mask_w + b  (WMMA path, weights pre-transposed)
    conv(rowv, mfeat, 20, mask_b, N, 256, 8192, 1, 1, 1, 0);

    // ================= decoder ==============================================
    k_mfeat_cat<<<(N * 32 * 256 + 255) / 256, 256, 0, stream>>>(mfeat, cat4, N);
    k_crop_cat<<<(N * 512 * 256 + 255) / 256, 256, 0, stream>>>(keep4, cat4, pos, N, 512, 32, 1, 544, 32);
    conv(cat4, A4, 11, Bp(11), N, 544, 512, 16, 16, 3, 1);
    conv(A4, B4, 12, Bp(12), N, 512, 512, 16, 16, 3, 1);

    k_up2_cat<<<(N * 512 * 1024 + 255) / 256, 256, 0, stream>>>(B4, cat3, N, 512, 16, 16, 768);
    k_crop_cat<<<(N * 256 * 1024 + 255) / 256, 256, 0, stream>>>(keep3, cat3, pos, N, 256, 64, 2, 768, 512);
    conv(cat3, A3, 13, Bp(13), N, 768, 256, 32, 32, 3, 1);
    conv(A3, B3, 14, Bp(14), N, 256, 256, 32, 32, 3, 1);

    k_up2_cat<<<(N * 256 * 4096 + 255) / 256, 256, 0, stream>>>(B3, cat2, N, 256, 32, 32, 384);
    k_crop_cat<<<(N * 128 * 4096 + 255) / 256, 256, 0, stream>>>(keep2, cat2, pos, N, 128, 128, 4, 384, 256);
    conv(cat2, A2, 15, Bp(15), N, 384, 128, 64, 64, 3, 1);
    conv(A2, B2, 16, Bp(16), N, 128, 128, 64, 64, 3, 1);

    k_up2_cat<<<(N * 128 * 16384 + 255) / 256, 256, 0, stream>>>(B2, cat1, N, 128, 64, 64, 192);
    k_crop_cat<<<(N * 64 * 16384 + 255) / 256, 256, 0, stream>>>(keep1, cat1, pos, N, 64, 256, 8, 192, 128);
    conv(cat1, A1, 17, Bp(17), N, 192, 64, 128, 128, 3, 1);
    conv(A1, B1, 18, Bp(18), N, 64, 64, 128, 128, 3, 1);
    conv(B1, fin, 19, Bp(19), N, 64, 1, 128, 128, 1, 0);

    // ================= outputs ==============================================
    float* out_f = (float*)d_out;
    k_out_score<<<(Bsz * Tn * 289 + 255) / 256, 256, 0, stream>>>(score, out_f, Bsz, Tn);
    k_out_mask<<<(Bsz * Tn * 16384 + 255) / 256, 256, 0, stream>>>(fin, out_f + Bsz * Tn * 289, Bsz, Tn);
}
